// UniGCNII_78735340470817
// MI455X (gfx1250) — matrix operand order repaired and checked by reading, not compiled, not run on hardware
//
#include <hip/hip_runtime.h>
#include <math.h>

#define N_V    100000
#define M_E    200000
#define NNZI   1600000
#define NFEAT  128
#define NHID   64
#define NCLASS 40
#define NLAYER 8
#define ALPHA  0.1f
#define LAMDA  0.5f

typedef __attribute__((ext_vector_type(16))) _Float16 v16h;
typedef __attribute__((ext_vector_type(8)))  float    v8f;

__device__ __forceinline__ void atomic_add_f32(float* p, float v) {
    __hip_atomic_fetch_add(p, v, __ATOMIC_RELAXED, __HIP_MEMORY_SCOPE_AGENT);
}

// ---------------- zero fill (graph-capture safe, no memset API) ----------------
__global__ void k_zero(float4* __restrict__ p, long n4) {
    long i = (long)blockIdx.x * blockDim.x + threadIdx.x;
    long stride = (long)gridDim.x * blockDim.x;
    float4 z = {0.f, 0.f, 0.f, 0.f};
    for (; i < n4; i += stride) p[i] = z;
}

// ---------------- hyperedge incidence counts ----------------
__global__ void k_count(const int* __restrict__ edges, float* __restrict__ cnt, int nnz) {
    int i = blockIdx.x * blockDim.x + threadIdx.x;
    if (i < nnz) atomic_add_f32(&cnt[edges[i]], 1.0f);
}

// scaleE[e] = degE[e] / count[e]   (inv_counts * degE folded)
__global__ void k_scale_e(const float* __restrict__ cnt, const float* __restrict__ degE,
                          float* __restrict__ scaleE, int m) {
    int i = blockIdx.x * blockDim.x + threadIdx.x;
    if (i < m) {
        float c = cnt[i];
        scaleE[i] = (c > 0.f) ? (degE[i] / c) : 0.f;
    }
}

// ---------------- sparse scatters (float4 gather + hw f32 atomics) ----------------
// vertex -> edge: Xe[e] += h[v]
__global__ void k_scatter_v2e(const float4* __restrict__ h, const int* __restrict__ vertex,
                              const int* __restrict__ edges, float* __restrict__ Xe, int nnz) {
    int tid = blockIdx.x * blockDim.x + threadIdx.x;
    int j = tid >> 4;               // incidence
    if (j >= nnz) return;
    int c = tid & 15;               // float4 chunk within 64-float row
    int v = vertex[j], e = edges[j];
    float4 val = h[(long)v * 16 + c];
    float* dst = Xe + (long)e * 64 + c * 4;
    atomic_add_f32(dst + 0, val.x);
    atomic_add_f32(dst + 1, val.y);
    atomic_add_f32(dst + 2, val.z);
    atomic_add_f32(dst + 3, val.w);
}

// edge -> vertex: Xv[v] += Xe[e] * scaleE[e]
__global__ void k_scatter_e2v(const float4* __restrict__ Xe, const float* __restrict__ scaleE,
                              const int* __restrict__ vertex, const int* __restrict__ edges,
                              float* __restrict__ Xv, int nnz) {
    int tid = blockIdx.x * blockDim.x + threadIdx.x;
    int j = tid >> 4;
    if (j >= nnz) return;
    int c = tid & 15;
    int v = vertex[j], e = edges[j];
    float s = scaleE[e];
    float4 val = Xe[(long)e * 16 + c];
    float* dst = Xv + (long)v * 64 + c * 4;
    atomic_add_f32(dst + 0, val.x * s);
    atomic_add_f32(dst + 1, val.y * s);
    atomic_add_f32(dst + 2, val.z * s);
    atomic_add_f32(dst + 3, val.w * s);
}

// ---------------- fused degV-scale + L2 row-normalize + alpha residual mix ----------------
// Xv <- (1-a) * normalize(Xv * degV) + a * h0     (one wave32 per row, shfl_xor reduce)
__global__ void k_norm_mix(float* __restrict__ Xv, const float* __restrict__ degV,
                           const float* __restrict__ h0, int n) {
    int row  = blockIdx.x * (blockDim.x >> 5) + (threadIdx.x >> 5);
    if (row >= n) return;
    int lane = threadIdx.x & 31;
    float2*       rp = (float2*)(Xv + (long)row * 64);
    const float2* hp = (const float2*)(h0 + (long)row * 64);
    float dv = degV[row];
    float2 v = rp[lane];
    v.x *= dv; v.y *= dv;
    float ss = v.x * v.x + v.y * v.y;
#pragma unroll
    for (int m = 16; m >= 1; m >>= 1) ss += __shfl_xor(ss, m, 32);
    float inv = (ss > 0.f) ? rsqrtf(ss) : 0.f;
    float2 h0v = hp[lane];
    float2 xi;
    xi.x = (1.f - ALPHA) * v.x * inv + ALPHA * h0v.x;
    xi.y = (1.f - ALPHA) * v.y * inv + ALPHA * h0v.y;
    rp[lane] = xi;
}

// ---------------- weight pre-pack into per-lane WMMA B fragments (f16) ----------------
// Fragment layout: element index ((chunk * NT + wv) * 32 + lane) holds the v16h for that
// (K-chunk, col-tile, lane): lane covers col = wv*16 + (lane&15), K = chunk*32 + hi*16 + j.
// Folds Weff = wmul*W + iadd*I and zero-pads columns >= NCOLS_REAL.
template<int KDIM, int NCOLS, int NCOLS_REAL, bool FOLD_IDENT>
__global__ void k_pack_b(const float* __restrict__ W, v16h* __restrict__ out,
                         float wmul, float iadd) {
    constexpr int NT = NCOLS / 16;
    constexpr int NCH = KDIM / 32;
    int tid = blockIdx.x * blockDim.x + threadIdx.x;
    if (tid >= NCH * NT * 32) return;
    int lane = tid & 31;
    int t    = tid >> 5;
    int wv   = t % NT;
    int ch   = t / NT;
    int hi   = lane >> 4;
    int ncol = wv * 16 + (lane & 15);
    int  ncolc = (ncol < NCOLS_REAL) ? ncol : NCOLS_REAL - 1;
    bool colok = (ncol < NCOLS_REAL);
    int kb = ch * 32 + hi * 16;
    v16h b;
#pragma unroll
    for (int j = 0; j < 16; ++j) {
        float w = W[(kb + j) * NCOLS_REAL + ncolc];
        if (!colok) w = 0.f;
        if (FOLD_IDENT) w = wmul * w + (((kb + j) == ncol) ? iadd : 0.f);
        b[j] = (_Float16)w;
    }
    out[tid] = b;
}

// ---------------- WMMA GEMM: 16-row tile per block, 16-col tile per wave ----------------
// out = post( A[rows,KDIM] @ Bpack ) ; +bias ; relu.  B fragments are pre-packed f16 ->
// one contiguous 32B load per fragment; A is converted f32->f16 in registers.
template<int KDIM, int NCOLS, int NCOLS_REAL, bool RELU, bool HAS_BIAS, bool DUAL_STORE>
__global__ void k_gemm(const float* __restrict__ A, const v16h* __restrict__ Bp,
                       const float* __restrict__ bias, float* __restrict__ out,
                       float* __restrict__ out2) {
    constexpr int NT = NCOLS / 16;
    int lane  = threadIdx.x;                 // 0..31
    int wv    = threadIdx.y;                 // column tile index
    int tileM = blockIdx.x * 16;
    int hi    = lane >> 4;                   // half-wave selector (0/1)
    int mrow  = tileM + (lane & 15);
    int ncol  = wv * 16 + (lane & 15);
    int  ncolc = (NCOLS <= NCOLS_REAL) ? ncol
                                       : (ncol < NCOLS_REAL ? ncol : NCOLS_REAL - 1);
    bool colok = (NCOLS <= NCOLS_REAL) ? true : (ncol < NCOLS_REAL);

    const float* arow = A + (long)mrow * KDIM + hi * 16;  // A[mrow][hi*16 + ...]
    const v16h*  bp   = Bp + wv * 32 + lane;              // fragment base for this lane

    v8f acc = {0.f, 0.f, 0.f, 0.f, 0.f, 0.f, 0.f, 0.f};

#pragma unroll
    for (int ch = 0; ch < KDIM / 32; ++ch) {
        // A fragment: 16x32 f16, lane holds row mrow, K = ch*32 + hi*16 + {0..15}
        v16h a;
#pragma unroll
        for (int j = 0; j < 8; ++j) {
            a[j]     = (_Float16)arow[ch * 32 + j];
            a[j + 8] = (_Float16)arow[ch * 32 + 8 + j];
        }
        v16h b = bp[ch * NT * 32];                        // 32B contiguous fragment load
        acc = __builtin_amdgcn_wmma_f32_16x16x32_f16(
            /*neg_a=*/false, a, /*neg_b=*/false, b,
            /*c_mod=*/(short)0, acc, /*reuse_a=*/false, /*reuse_b=*/false);
    }

    int mbase = tileM + hi * 8;              // D layout: VGPR r -> row mbase+r, col = lane&15
    float bval = 0.f;
    if (HAS_BIAS) {
        bval = bias[ncolc];
        if (NCOLS > NCOLS_REAL) bval = colok ? bval : 0.f;
    }
#pragma unroll
    for (int r = 0; r < 8; ++r) {
        float v = acc[r] + bval;
        if (RELU) v = fmaxf(v, 0.f);
        if (colok) {
            long idx = (long)(mbase + r) * NCOLS_REAL + ncol;
            out[idx] = v;
            if (DUAL_STORE) out2[idx] = v;
        }
    }
}

extern "C" void kernel_launch(void* const* d_in, const int* in_sizes, int n_in,
                              void* d_out, int out_size, void* d_ws, size_t ws_size,
                              hipStream_t stream) {
    const float* x      = (const float*)d_in[0];
    const int*   vertex = (const int*)d_in[1];
    const int*   edges  = (const int*)d_in[2];
    const float* degE   = (const float*)d_in[3];
    const float* degV   = (const float*)d_in[4];
    const float* W0     = (const float*)d_in[5];
    const float* b0     = (const float*)d_in[6];
    const float* Ws     = (const float*)d_in[7];
    const float* Wout   = (const float*)d_in[8];
    const float* bout   = (const float*)d_in[9];
    float* out = (float*)d_out;

    // workspace layout (floats, then 32B-aligned v16h pack region)
    float* ws = (float*)d_ws;
    float* h      = ws;                          // N*64
    float* h0     = h  + (long)N_V * NHID;       // N*64
    float* Xe     = h0 + (long)N_V * NHID;       // M*64
    float* Xv     = Xe + (long)M_E * NHID;       // N*64
    float* scaleE = Xv + (long)N_V * NHID;       // M   (counts computed in place)
    v16h* packW0   = (v16h*)(scaleE + M_E);      // 4 chunks * 4 tiles * 32 = 512 frags
    v16h* packWs   = packW0 + 512;               // 8 layers * 2*4*32 = 2048 frags
    v16h* packWout = packWs + 8 * 256;           // 2 chunks * 3 tiles * 32 = 192 frags

    // 0) pre-pack all weights into WMMA B-fragment layout (tiny single-block kernels)
    k_pack_b<NFEAT, NHID, NHID, false><<<1, 512, 0, stream>>>(W0, packW0, 1.f, 0.f);
    for (int i = 0; i < NLAYER; ++i) {
        float beta = logf(LAMDA / (float)(i + 1) + 1.0f);
        k_pack_b<NHID, NHID, NHID, true><<<1, 256, 0, stream>>>(
            Ws + (long)i * NHID * NHID, packWs + i * 256, beta, 1.f - beta);
    }
    k_pack_b<NHID, 48, NCLASS, false><<<1, 192, 0, stream>>>(Wout, packWout, 1.f, 0.f);

    // 1) per-edge scale = degE / count
    k_zero<<<256, 256, 0, stream>>>((float4*)scaleE, (long)M_E / 4);
    k_count<<<(NNZI + 255) / 256, 256, 0, stream>>>(edges, scaleE, NNZI);
    k_scale_e<<<(M_E + 255) / 256, 256, 0, stream>>>(scaleE, degE, scaleE, M_E);

    // 2) input GEMM: h = h0 = relu(x @ W0 + b0)
    dim3 gb(32, NHID / 16);                      // 4 waves, one 16-col tile each
    k_gemm<NFEAT, NHID, NHID, true, true, true><<<N_V / 16, gb, 0, stream>>>(
        x, packW0, b0, h, h0);

    // 3) propagation layers
    for (int i = 0; i < NLAYER; ++i) {
        k_zero<<<2048, 256, 0, stream>>>((float4*)Xe, (long)M_E * NHID / 4);
        k_scatter_v2e<<<(NNZI * 16) / 256, 256, 0, stream>>>(
            (const float4*)h, vertex, edges, Xe, NNZI);
        k_zero<<<2048, 256, 0, stream>>>((float4*)Xv, (long)N_V * NHID / 4);
        k_scatter_e2v<<<(NNZI * 16) / 256, 256, 0, stream>>>(
            (const float4*)Xe, scaleE, vertex, edges, Xv, NNZI);
        k_norm_mix<<<N_V / 8, 256, 0, stream>>>(Xv, degV, h0, N_V);
        // h = relu(Xi @ ((1-beta)I + beta*Ws[i]))  (identity folded into pack)
        k_gemm<NHID, NHID, NHID, true, false, false><<<N_V / 16, gb, 0, stream>>>(
            Xv, packWs + i * 256, nullptr, h, nullptr);
    }

    // 4) output GEMM: out = h @ Wout + bout   (40 real cols, 48 padded -> 3 col tiles)
    dim3 go(32, 3);
    k_gemm<NHID, 48, NCLASS, false, true, false><<<N_V / 16, go, 0, stream>>>(
        h, packWout, bout, out, nullptr);
}
